// MultiHeadSelfAttention_89979564851698
// MI455X (gfx1250) — matrix operand order repaired and checked by reading, not compiled, not run on hardware
//
#include <hip/hip_runtime.h>
#include <hip/hip_bf16.h>

#define B_   8
#define T_   2048
#define C_   256
#define H_   4
#define D_   64
#define NQKV (3 * C_)      // 768
#define BT_  (B_ * T_)     // 16384

// LDS tile strides (bf16 elements), padded for conflict-free ds_load_b128
#define LKS  72            // K tile row stride  (32 rows x 64 valid)
#define LVS  40            // V^T tile row stride (64 rows x 32 valid)

typedef __attribute__((ext_vector_type(16))) __bf16 v16bf;
typedef __attribute__((ext_vector_type(8)))  __bf16 v8bf;
typedef __attribute__((ext_vector_type(8)))  float  v8f;
typedef __attribute__((ext_vector_type(4)))  int    v4i;

#define AS1 __attribute__((address_space(1)))
#define AS3 __attribute__((address_space(3)))

#if __has_builtin(__builtin_amdgcn_global_load_async_to_lds_b128) && \
    __has_builtin(__builtin_amdgcn_s_wait_asynccnt)
#define USE_ASYNC_LDS 1
#else
#define USE_ASYNC_LDS 0
#endif

// ---------------------------------------------------------------- WMMA helpers

__device__ __forceinline__ v8f wmma_bf16(v16bf a, v16bf b, v8f c) {
  return __builtin_amdgcn_wmma_f32_16x16x32_bf16(
      /*neg_a=*/false, a, /*neg_b=*/false, b,
      /*c_mod=*/(short)0, c, /*reuse_a=*/false, /*reuse_b=*/false);
}

// A-operand (16x32 bf16): lane = row (lane&15); elems e <-> K = 16*lhi + e.
__device__ __forceinline__ v16bf load_afrag(const __bf16* p, int lhi) {
  union { v16bf v; v8bf h[2]; } u;
  u.h[0] = *(const v8bf*)(p + 16 * lhi);
  u.h[1] = *(const v8bf*)(p + 16 * lhi + 8);
  return u.v;
}

// B-operand (32x16 bf16): lane = column (lane&15); elems e <-> K = 16*lhi + e.
__device__ __forceinline__ v16bf load_bfrag(const __bf16* p, int lhi) {
  union { v16bf v; v8bf h[2]; } u;
  u.h[0] = *(const v8bf*)(p + 16 * lhi);
  u.h[1] = *(const v8bf*)(p + 16 * lhi + 8);
  return u.v;
}

// 16B global -> LDS copy (async DMA when available)
__device__ __forceinline__ void cp_b128(const __bf16* src, __bf16* ldst) {
#if USE_ASYNC_LDS
  __builtin_amdgcn_global_load_async_to_lds_b128(
      (AS1 v4i*)(v4i*)const_cast<__bf16*>(src), (AS3 v4i*)(v4i*)ldst, 0, 0);
#else
  *(v8bf*)ldst = *(const v8bf*)src;
#endif
}

// ---------------------------------------------------------------- conversions

__global__ void mhsa_cvt_bf16(const float* __restrict__ in, __bf16* __restrict__ out, int n) {
  int i = blockIdx.x * blockDim.x + threadIdx.x;
  if (i < n) out[i] = (__bf16)in[i];
}

// out[n*K + k] = (bf16) in[k*N + n]
__global__ void mhsa_transpose_bf16(const float* __restrict__ in, __bf16* __restrict__ out,
                                    int K, int N) {
  int i = blockIdx.x * blockDim.x + threadIdx.x;
  if (i < K * N) {
    int n = i / K, k = i - n * K;
    out[i] = (__bf16)in[(size_t)k * N + n];
  }
}

// ---------------------------------------------------------------- QKV GEMM

__global__ void mhsa_qkv_gemm(const __bf16* __restrict__ xbf,
                              const __bf16* __restrict__ wT,   // (NQKV, C_)
                              const float*  __restrict__ bias, // (NQKV)
                              __bf16* __restrict__ Qbf,
                              __bf16* __restrict__ Kbf,
                              __bf16* __restrict__ Vt) {
  const int lane = threadIdx.x & 31;
  const int wave = threadIdx.x >> 5;
  const int lhi  = lane >> 4;
  const int lm   = lane & 15;

  const int tiles_n = NQKV / 16;                 // 48
  const int tile    = blockIdx.x * 8 + wave;
  const int tm = tile / tiles_n, tn = tile - tm * tiles_n;
  const int m0 = tm * 16, n0 = tn * 16;

  const __bf16* arow = xbf + (size_t)(m0 + lm) * C_;
  const __bf16* brow = wT  + (size_t)(n0 + lm) * C_;

  v8f c = {};
#pragma unroll
  for (int k0 = 0; k0 < C_; k0 += 32) {
    c = wmma_bf16(load_afrag(arow + k0, lhi), load_bfrag(brow + k0, lhi), c);
  }

  const int gn    = n0 + lm;
  const float bv  = bias[gn];
  const int which = gn / C_;               // 0=q 1=k 2=v (wave-uniform)
  const int cc    = gn - which * C_;
  const int hh    = cc >> 6;
  const int dd    = cc & 63;
  const int bb    = m0 / T_;
  const int t0    = (m0 - bb * T_) + 8 * lhi;
  const size_t headQK = ((size_t)bb * H_ + hh) * T_;
  const size_t headV  = (((size_t)bb * H_ + hh) * D_ + dd) * T_;

#pragma unroll
  for (int r = 0; r < 8; ++r) {
    const int t = t0 + r;
    const float v = c[r] + bv;
    if (which == 0)      Qbf[(headQK + t) * D_ + dd] = (__bf16)(v * 0.125f); // 1/sqrt(64)
    else if (which == 1) Kbf[(headQK + t) * D_ + dd] = (__bf16)v;
    else                 Vt [headV + t]              = (__bf16)v;
  }
}

// ---------------------------------------------------------------- flash attention

// Stage one 32-kv tile into LDS: K (32x64) and V^T (64x32), 2 async ops/wave.
__device__ __forceinline__ void stage_tile(const __bf16* __restrict__ Kh,
                                           const __bf16* __restrict__ Vh,
                                           int kv, __bf16* lk, __bf16* lv, int tid) {
  { // K: 256 threads x 16B
    const int row = tid >> 3, c8 = (tid & 7) * 8;
    cp_b128(Kh + (size_t)(kv + row) * D_ + c8, lk + row * LKS + c8);
  }
  { // V^T: 256 threads x 16B
    const int row = tid >> 2, c8 = (tid & 3) * 8;
    cp_b128(Vh + (size_t)row * T_ + kv + c8, lv + row * LVS + c8);
  }
}

// grid = (B_*H_, T_/128), 256 threads (8 waves); each wave owns 16 q rows.
// K/V tiles staged once per block into double-buffered LDS via async DMA.
__global__ void __launch_bounds__(256) mhsa_flash(const __bf16* __restrict__ Qbf,
                                                  const __bf16* __restrict__ Kbf,
                                                  const __bf16* __restrict__ Vt,
                                                  __bf16* __restrict__ Obf) {
  __shared__ __bf16 lk[2][32 * LKS];
  __shared__ __bf16 lv[2][64 * LVS];

  const int tid  = threadIdx.x;
  const int lane = tid & 31;
  const int wave = tid >> 5;
  const int lhi  = lane >> 4;
  const int lm   = lane & 15;

  const int bh = blockIdx.x;
  const int bb = bh >> 2, hh = bh & (H_ - 1);
  const int qb_blk = blockIdx.y * 128;
  const int qb = qb_blk + wave * 16;             // this wave's q-row base

  const __bf16* Qh = Qbf + (size_t)bh * T_ * D_;
  const __bf16* Kh = Kbf + (size_t)bh * T_ * D_;
  const __bf16* Vh = Vt  + (size_t)bh * D_ * T_;

  // Q^T B-fragments (lane = q row m), register-resident across all kv tiles
  const __bf16* qrow = Qh + (size_t)(qb + lm) * D_;
  const v16bf qt0 = load_bfrag(qrow + 0,  lhi);
  const v16bf qt1 = load_bfrag(qrow + 32, lhi);

  v8f ot[4] = {{}, {}, {}, {}};
  float m_i = -1e30f, l_i = 0.0f;

  const int ntiles = qb_blk / 32 + 4;            // block-uniform causal trip count

  stage_tile(Kh, Vh, 0, lk[0], lv[0], tid);      // prologue DMA

  for (int it = 0; it < ntiles; ++it) {
    const int kv  = it * 32;
    const int cur = it & 1;

    if (it + 1 < ntiles) {                        // overlap next tile's DMA
      stage_tile(Kh, Vh, kv + 32, lk[cur ^ 1], lv[cur ^ 1], tid);
#if USE_ASYNC_LDS
      __builtin_amdgcn_s_wait_asynccnt(2);        // current tile's 2 ops retired
#endif
    } else {
#if USE_ASYNC_LDS
      __builtin_amdgcn_s_wait_asynccnt(0);
#endif
    }
    __syncthreads();                              // tile `it` fully in LDS

    if (kv <= qb + 15) {                          // wave-uniform causal skip
      // S^T (kv x m): A = K rows from LDS, B = Q^T; chain over d = 0..63
      const __bf16* lkc = lk[cur];
      v8f st0 = {}, st1 = {};
      st0 = wmma_bf16(load_afrag(lkc + lm * LKS + 0,  lhi), qt0, st0);
      st0 = wmma_bf16(load_afrag(lkc + lm * LKS + 32, lhi), qt1, st0);
      st1 = wmma_bf16(load_afrag(lkc + (16 + lm) * LKS + 0,  lhi), qt0, st1);
      st1 = wmma_bf16(load_afrag(lkc + (16 + lm) * LKS + 32, lhi), qt1, st1);

      const int mg = qb + lm;
      if (kv + 31 > qb) {                         // diagonal tile: mask n > m
        const int nb0 = kv + 8 * lhi, nb1 = kv + 16 + 8 * lhi;
#pragma unroll
        for (int r = 0; r < 8; ++r) {
          if (nb0 + r > mg) st0[r] = -1e30f;
          if (nb1 + r > mg) st1[r] = -1e30f;
        }
      }

      // online softmax: lane-local + one xor-16 combine
      float tmax = -1e30f;
#pragma unroll
      for (int r = 0; r < 8; ++r) tmax = fmaxf(tmax, fmaxf(st0[r], st1[r]));
      tmax = fmaxf(tmax, __shfl_xor(tmax, 16, 32));
      const float mn   = fmaxf(m_i, tmax);
      const float corr = __expf(m_i - mn);

      float p0[8], p1[8], ls = 0.0f;
#pragma unroll
      for (int r = 0; r < 8; ++r) {
        p0[r] = __expf(st0[r] - mn);
        p1[r] = __expf(st1[r] - mn);
        ls += p0[r] + p1[r];
      }
      ls += __shfl_xor(ls, 16, 32);
      l_i = l_i * corr + ls;
      m_i = mn;
#pragma unroll
      for (int t = 0; t < 4; ++t)
#pragma unroll
        for (int r = 0; r < 8; ++r) ot[t][r] *= corr;

      // P^T as B-operand: one half-swap shuffle per register pair
      v16bf pf;
#pragma unroll
      for (int r = 0; r < 8; ++r) {
        const float o0 = __shfl_xor(p0[r], 16, 32);
        const float o1 = __shfl_xor(p1[r], 16, 32);
        pf[r]     = (__bf16)(lhi ? o1 : p0[r]);
        pf[8 + r] = (__bf16)(lhi ? p1[r] : o0);
      }

      // O^T(d, m) += V^T(d, kv) * P^T(kv, m), four 16-wide d-blocks from LDS
#pragma unroll
      for (int t = 0; t < 4; ++t)
        ot[t] = wmma_bf16(load_afrag(lv[cur] + (t * 16 + lm) * LVS, lhi), pf, ot[t]);
    }

    __syncthreads();                              // protect buffer before re-stage
  }

  // finalize: scale O^T column m by 1/l[m], store bf16 (B,T,C)
  const float inv = 1.0f / l_i;
  __bf16* orow = Obf + ((size_t)bb * T_ + (qb + lm)) * C_ + hh * D_;
#pragma unroll
  for (int t = 0; t < 4; ++t) {
    const int dbase = t * 16 + 8 * lhi;
#pragma unroll
    for (int r = 0; r < 8; r += 2) {
      union { __bf16 h[2]; unsigned u; } pk;
      pk.h[0] = (__bf16)(ot[t][r]     * inv);
      pk.h[1] = (__bf16)(ot[t][r + 1] * inv);
      *(unsigned*)(orow + dbase + r) = pk.u;
    }
  }
}

// ---------------------------------------------------------------- out projection

__global__ void mhsa_proj_gemm(const __bf16* __restrict__ Obf,
                               const __bf16* __restrict__ wT,   // Wproj^T (C_, C_)
                               const float*  __restrict__ bias,
                               float* __restrict__ out) {
  const int lane = threadIdx.x & 31;
  const int wave = threadIdx.x >> 5;
  const int lhi  = lane >> 4;
  const int lm   = lane & 15;

  const int tiles_n = C_ / 16;                   // 16
  const int tile = blockIdx.x * 8 + wave;
  const int tm = tile / tiles_n, tn = tile - tm * tiles_n;
  const int m0 = tm * 16, n0 = tn * 16;

  const __bf16* arow = Obf + (size_t)(m0 + lm) * C_;
  const __bf16* brow = wT  + (size_t)(n0 + lm) * C_;

  v8f c = {};
#pragma unroll
  for (int k0 = 0; k0 < C_; k0 += 32)
    c = wmma_bf16(load_afrag(arow + k0, lhi), load_bfrag(brow + k0, lhi), c);

  const int gn = n0 + lm;
  const float bv = bias[gn];
  float* od = out + (size_t)(m0 + 8 * lhi) * C_ + gn;
#pragma unroll
  for (int r = 0; r < 8; ++r) od[(size_t)r * C_] = c[r] + bv;
}

// ---------------------------------------------------------------- launch

extern "C" void kernel_launch(void* const* d_in, const int* in_sizes, int n_in,
                              void* d_out, int out_size, void* d_ws, size_t ws_size,
                              hipStream_t stream) {
  const float* x     = (const float*)d_in[0];   // (B,T,C)
  const float* Wqkv  = (const float*)d_in[1];   // (C, 3C)
  const float* bqkv  = (const float*)d_in[2];   // (3C)
  const float* Wproj = (const float*)d_in[3];   // (C, C)
  const float* bproj = (const float*)d_in[4];   // (C)
  float* out = (float*)d_out;

  char* ws = (char*)d_ws;
  size_t off = 0;
  auto take = [&](size_t bytes) -> void* {
    void* p = ws + off;
    off += (bytes + 255) & ~(size_t)255;
    return p;
  };

  __bf16* xbf   = (__bf16*)take((size_t)BT_ * C_ * 2);
  __bf16* wqkvT = (__bf16*)take((size_t)NQKV * C_ * 2);
  __bf16* wprjT = (__bf16*)take((size_t)C_ * C_ * 2);
  __bf16* Qbf   = (__bf16*)take((size_t)BT_ * C_ * 2);     // (B,H,T,D)
  __bf16* Kbf   = (__bf16*)take((size_t)BT_ * C_ * 2);     // (B,H,T,D)
  __bf16* Vt    = (__bf16*)take((size_t)BT_ * C_ * 2);     // (B,H,D,T)
  __bf16* Obf   = (__bf16*)take((size_t)BT_ * C_ * 2);     // (B,T,C)

  {
    int n = BT_ * C_;
    mhsa_cvt_bf16<<<(n + 255) / 256, 256, 0, stream>>>(x, xbf, n);
    int nw = NQKV * C_;
    mhsa_transpose_bf16<<<(nw + 255) / 256, 256, 0, stream>>>(Wqkv, wqkvT, C_, NQKV);
    int np = C_ * C_;
    mhsa_transpose_bf16<<<(np + 255) / 256, 256, 0, stream>>>(Wproj, wprjT, C_, C_);
  }
  {
    int tiles = (BT_ / 16) * (NQKV / 16);
    mhsa_qkv_gemm<<<tiles / 8, 256, 0, stream>>>(xbf, wqkvT, bqkv, Qbf, Kbf, Vt);
  }
  {
    dim3 grid(B_ * H_, T_ / 128);
    mhsa_flash<<<grid, 256, 0, stream>>>(Qbf, Kbf, Vt, Obf);
  }
  {
    int tiles = (BT_ / 16) * (C_ / 16);
    mhsa_proj_gemm<<<tiles / 8, 256, 0, stream>>>(Obf, wprjT, bproj, out);
  }
}
